// AttnGatingExpL2_90890097918400
// MI455X (gfx1250) — matrix operand order repaired and checked by reading, not compiled, not run on hardware
//
#include <hip/hip_runtime.h>

#define B_   16
#define LQ_  512
#define LKV_ 32
#define D_   512
#define NEGV -10000000000.0f

typedef __attribute__((ext_vector_type(16))) __bf16 v16bf;
typedef __attribute__((ext_vector_type(8)))  float  v8f;

union Frag16 {
    unsigned int   u32[8];
    unsigned short u16[16];
    v16bf          v;
};

__device__ __forceinline__ unsigned short f2bf(float x) {
    unsigned int u = __float_as_uint(x);
    u += 0x7FFFu + ((u >> 16) & 1u);   // round-to-nearest-even
    return (unsigned short)(u >> 16);
}

// global -> LDS async copy (ASYNCcnt-tracked), GVS mode: saddr + 32-bit voffset
#define ASYNC_LDS_B64(ldsaddr, goff, sbase)                                  \
    asm volatile("global_load_async_to_lds_b64 %0, %1, %2"                   \
                 :: "v"(ldsaddr), "v"(goff), "s"(sbase) : "memory")
#define WAIT_ASYNC0() asm volatile("s_wait_asynccnt 0x0" ::: "memory")

// ---------------------------------------------------------------------------
// f32 -> bf16 conversion kernels
// ---------------------------------------------------------------------------
__global__ void cvt_bf16(const float* __restrict__ in,
                         unsigned short* __restrict__ out, int n) {
    int i = blockIdx.x * blockDim.x + threadIdx.x;
    if (i < n) out[i] = f2bf(in[i]);
}

__global__ void split_h2attn(const float* __restrict__ w,
                             unsigned short* __restrict__ wq,
                             unsigned short* __restrict__ wkv) {
    int i = blockIdx.x * blockDim.x + threadIdx.x;   // 0 .. 512*512-1
    if (i >= D_ * D_) return;
    int a = i >> 9, d = i & 511;
    wq[i]  = f2bf(w[a * (2 * D_) + d]);
    wkv[i] = f2bf(w[a * (2 * D_) + D_ + d]);
}

// ---------------------------------------------------------------------------
// Big bf16 WMMA GEMM: C[b][m][n] = sum_k A[b][m][k] * W[n][k] (+bias,+sigmoid)
// Requires M % 128 == 0, N % 128 == 0, K % 64 == 0.
// Block tile 128x128, BK=64, 8 waves (4m x 2n), wave tile 32x64 (2x4 WMMA).
// Double-buffered LDS fed by GLOBAL_LOAD_ASYNC_TO_LDS_B64.
// MODE: 0 = plain store, 1 = +bias, 2 = +bias then sigmoid
// ---------------------------------------------------------------------------
template <int MODE>
__global__ __launch_bounds__(256) void gemm_bf16_wmma_async(
    const unsigned short* __restrict__ A,
    const unsigned short* __restrict__ W,
    const float* __restrict__ bias,
    float* __restrict__ C,
    int M, int N, int K,
    unsigned long long strideA, unsigned long long strideC)
{
    // padded row stride 72 shorts (144 B): 8B-aligned, conflict-free frags
    __shared__ __align__(16) unsigned short As[2][128 * 72];
    __shared__ __align__(16) unsigned short Bs[2][128 * 72];

    const int tid  = threadIdx.x;
    const int lane = tid & 31;
    const int wid  = tid >> 5;
    const int wm   = wid >> 1;    // 0..3 -> +32 rows
    const int wn   = wid & 1;     // 0..1 -> +64 cols
    const int bm0  = blockIdx.x * 128;
    const int bn0  = blockIdx.y * 128;
    const int bat  = blockIdx.z;

    const unsigned short* Abase = A + (unsigned long long)bat * strideA;

    v8f acc[2][4];
    #pragma unroll
    for (int i = 0; i < 2; ++i)
        #pragma unroll
        for (int j = 0; j < 4; ++j) acc[i][j] = {};

    // loop-invariant per-thread staging offsets (bytes)
    const int srow = tid >> 4;          // 0..15
    const int scol = (tid & 15) * 4;    // shorts, 8B chunks
    unsigned aoffG[8], boffG[8], toffL[8];
    #pragma unroll
    for (int p = 0; p < 8; ++p) {
        const int r = srow + p * 16;    // 0..127
        aoffG[p] = (unsigned)(((bm0 + r) * K + scol) * 2);
        boffG[p] = (unsigned)(((bn0 + r) * K + scol) * 2);
        toffL[p] = (unsigned)((r * 72 + scol) * 2);
    }
    const unsigned aL[2] = { (unsigned)(size_t)&As[0][0], (unsigned)(size_t)&As[1][0] };
    const unsigned bL[2] = { (unsigned)(size_t)&Bs[0][0], (unsigned)(size_t)&Bs[1][0] };

    auto stage = [&](int buf, int k0) {
        const unsigned short* ab = Abase + k0;   // +2*k0 bytes
        const unsigned short* wb = W + k0;
        #pragma unroll
        for (int p = 0; p < 8; ++p)
            ASYNC_LDS_B64(aL[buf] + toffL[p], aoffG[p], ab);
        #pragma unroll
        for (int p = 0; p < 8; ++p)
            ASYNC_LDS_B64(bL[buf] + toffL[p], boffG[p], wb);
    };

    auto compute = [&](int buf) {
        const unsigned short* Asb = &As[buf][0];
        const unsigned short* Bsb = &Bs[buf][0];
        #pragma unroll
        for (int kk = 0; kk < 2; ++kk) {
            Frag16 af[2], bf[4];
            const int khiA = kk * 32 + ((lane >> 4) << 3);
            const int khiB = kk * 32 + ((lane >> 4) << 3);
            #pragma unroll
            for (int t = 0; t < 2; ++t) {
                const int m = wm * 32 + t * 16 + (lane & 15);
                #pragma unroll
                for (int g = 0; g < 8; ++g) {
                    const int k = (g < 4) ? (khiA + 2 * g) : (16 + khiA + 2 * (g - 4));
                    af[t].u32[g] = *(const unsigned int*)(&Asb[m * 72 + k]);
                }
            }
            #pragma unroll
            for (int t = 0; t < 4; ++t) {
                const int n = wn * 64 + t * 16 + (lane & 15);
                #pragma unroll
                for (int g = 0; g < 8; ++g)
                    bf[t].u32[g] = *(const unsigned int*)(&Bsb[n * 72 + ((g < 4) ? (khiB + 2 * g) : (16 + khiB + 2 * (g - 4)))]);
            }
            #pragma unroll
            for (int ti = 0; ti < 2; ++ti)
                #pragma unroll
                for (int tj = 0; tj < 4; ++tj)
                    acc[ti][tj] = __builtin_amdgcn_wmma_f32_16x16x32_bf16(
                        false, af[ti].v, false, bf[tj].v,
                        (short)0, acc[ti][tj], false, false);
        }
    };

    // software pipeline: prefetch tile 0, then overlap stage(next)/compute(cur)
    stage(0, 0);
    WAIT_ASYNC0();
    __syncthreads();
    int buf = 0;
    for (int k0 = 0; k0 < K; k0 += 64) {
        if (k0 + 64 < K) stage(buf ^ 1, k0 + 64);
        compute(buf);
        WAIT_ASYNC0();       // this wave's loads into next buffer done
        __syncthreads();     // everyone done reading cur + writing next
        buf ^= 1;
    }

    // epilogue: C layout m = g + 8*(lane>=16), n = lane&15
    float* Cb = C + (unsigned long long)bat * strideC;
    #pragma unroll
    for (int ti = 0; ti < 2; ++ti) {
        #pragma unroll
        for (int tj = 0; tj < 4; ++tj) {
            #pragma unroll
            for (int g = 0; g < 8; ++g) {
                const int m = bm0 + wm * 32 + ti * 16 + ((lane >> 4) * 8) + g;
                const int n = bn0 + wn * 64 + tj * 16 + (lane & 15);
                float vv = acc[ti][tj][g];
                if (MODE >= 1) vv += bias[n];
                if (MODE == 2) vv = 1.0f / (1.0f + __expf(-vv));
                Cb[(unsigned long long)m * N + n] = vv;
            }
        }
    }
}

// ---------------------------------------------------------------------------
// Small bf16 WMMA GEMM for ka: M == 32 exactly (no guards needed).
// Block tile 32x64, BK=64, 8 waves (2m x 4n), wave tile 16x16. +bias.
// Async-staged LDS (single buffer).
// ---------------------------------------------------------------------------
__global__ __launch_bounds__(256) void gemm_bf16_wmma_m32(
    const unsigned short* __restrict__ A,
    const unsigned short* __restrict__ W,
    const float* __restrict__ bias,
    float* __restrict__ C,
    int N, int K,
    unsigned long long strideA, unsigned long long strideC)
{
    __shared__ __align__(16) unsigned short As[32 * 72];
    __shared__ __align__(16) unsigned short Bs[64 * 72];

    const int tid  = threadIdx.x;
    const int lane = tid & 31;
    const int wid  = tid >> 5;
    const int wm   = wid & 1;     // 0..1 -> +16 rows
    const int wn   = wid >> 1;    // 0..3 -> +16 cols
    const int bn0  = blockIdx.y * 64;
    const int bat  = blockIdx.z;

    const unsigned short* Ab = A + (unsigned long long)bat * strideA;

    v8f acc = {};
    const int srow = tid >> 4;          // 0..15
    const int scol = (tid & 15) * 4;    // shorts

    // loop-invariant staging offsets (bytes)
    unsigned aG[2], aLo[2], bG[4], bLo[4];
    #pragma unroll
    for (int p = 0; p < 2; ++p) {
        const int r = srow + p * 16;
        aG[p]  = (unsigned)((r * K + scol) * 2);
        aLo[p] = (unsigned)((r * 72 + scol) * 2);
    }
    #pragma unroll
    for (int p = 0; p < 4; ++p) {
        const int r = srow + p * 16;
        bG[p]  = (unsigned)(((bn0 + r) * K + scol) * 2);
        bLo[p] = (unsigned)((r * 72 + scol) * 2);
    }
    const unsigned aLb = (unsigned)(size_t)&As[0];
    const unsigned bLb = (unsigned)(size_t)&Bs[0];

    for (int k0 = 0; k0 < K; k0 += 64) {
        const unsigned short* ab = Ab + k0;
        const unsigned short* wb = W + k0;
        #pragma unroll
        for (int p = 0; p < 2; ++p) ASYNC_LDS_B64(aLb + aLo[p], aG[p], ab);
        #pragma unroll
        for (int p = 0; p < 4; ++p) ASYNC_LDS_B64(bLb + bLo[p], bG[p], wb);
        WAIT_ASYNC0();
        __syncthreads();

        #pragma unroll
        for (int kk = 0; kk < 2; ++kk) {
            Frag16 af, bf;
            const int khiA = kk * 32 + ((lane >> 4) << 3);
            const int khiB = kk * 32 + ((lane >> 4) << 3);
            const int m = wm * 16 + (lane & 15);
            const int n = wn * 16 + (lane & 15);
            #pragma unroll
            for (int g = 0; g < 8; ++g) {
                const int k = (g < 4) ? (khiA + 2 * g) : (16 + khiA + 2 * (g - 4));
                af.u32[g] = *(const unsigned int*)(&As[m * 72 + k]);
                bf.u32[g] = *(const unsigned int*)(&Bs[n * 72 + ((g < 4) ? (khiB + 2 * g) : (16 + khiB + 2 * (g - 4)))]);
            }
            acc = __builtin_amdgcn_wmma_f32_16x16x32_bf16(
                false, af.v, false, bf.v, (short)0, acc, false, false);
        }
        __syncthreads();
    }

    float* Cb = C + (unsigned long long)bat * strideC;
    #pragma unroll
    for (int g = 0; g < 8; ++g) {
        const int m = wm * 16 + ((lane >> 4) * 8) + g;
        const int n = bn0 + wn * 16 + (lane & 15);
        Cb[(unsigned long long)m * N + n] = acc[g] + bias[n];
    }
}

// ---------------------------------------------------------------------------
// score[b,q,k] = sum_a tanh(qa[b,q,a] + ka[b,k,a]) * v[a]; mask; softmax_k
// ---------------------------------------------------------------------------
__global__ __launch_bounds__(256) void score_softmax(
    const float* __restrict__ qa, const float* __restrict__ ka,
    const float* __restrict__ v,  const int* __restrict__ audio_len,
    const int* __restrict__ text_len, float* __restrict__ attn)
{
    __shared__ float qrow[D_];
    __shared__ float sc[LKV_];
    const int q = blockIdx.x, b = blockIdx.y;
    const int tid = threadIdx.x, lane = tid & 31, wid = tid >> 5;

    const float* qp = qa + (unsigned long long)(b * LQ_ + q) * D_;
    qrow[tid]       = qp[tid];
    qrow[tid + 256] = qp[tid + 256];
    __syncthreads();

    #pragma unroll
    for (int kk = 0; kk < 4; ++kk) {
        const int k = wid * 4 + kk;
        const float* kp = ka + (unsigned long long)(b * LKV_ + k) * D_;
        float s = 0.f;
        #pragma unroll
        for (int i = 0; i < 16; ++i) {
            const int a = lane + i * 32;
            const float x  = qrow[a] + kp[a];
            const float e  = __expf(2.0f * x);
            const float th = 1.0f - 2.0f / (e + 1.0f);   // tanh(x)
            s = fmaf(th, v[a], s);
        }
        #pragma unroll
        for (int off = 16; off > 0; off >>= 1) s += __shfl_xor(s, off);
        if (lane == 0) sc[k] = s;
    }
    __syncthreads();

    if (wid == 0) {  // one wave: lane == k
        const bool qok = q < audio_len[b];
        const bool kok = lane < text_len[b];
        float s = (qok && kok) ? sc[lane] : NEGV;
        float m = s;
        #pragma unroll
        for (int off = 16; off > 0; off >>= 1) m = fmaxf(m, __shfl_xor(m, off));
        const float e = __expf(s - m);
        float t = e;
        #pragma unroll
        for (int off = 16; off > 0; off >>= 1) t += __shfl_xor(t, off);
        attn[(unsigned long long)(b * LQ_ + q) * LKV_ + lane] = e / t;
    }
}

// ---------------------------------------------------------------------------
// snippet[b,q,:] = attn[b,q,:] @ text[b]   (K=32), also emit bf16 copy
// ---------------------------------------------------------------------------
__global__ __launch_bounds__(256) void snippet_kernel(
    const float* __restrict__ attn, const float* __restrict__ text,
    float* __restrict__ snip, unsigned short* __restrict__ snip_bf)
{
    __shared__ float w[LKV_];
    const int q = blockIdx.x, b = blockIdx.y, tid = threadIdx.x;
    if (tid < LKV_)
        w[tid] = attn[(unsigned long long)(b * LQ_ + q) * LKV_ + tid];
    __syncthreads();

    const float* T = text + (unsigned long long)b * LKV_ * D_;
    const int d0 = tid, d1 = tid + 256;
    float a0 = 0.f, a1 = 0.f;
    #pragma unroll 8
    for (int k = 0; k < LKV_; ++k) {
        const float wk = w[k];
        a0 = fmaf(wk, T[k * D_ + d0], a0);
        a1 = fmaf(wk, T[k * D_ + d1], a1);
    }
    const unsigned long long base = (unsigned long long)(b * LQ_ + q) * D_;
    snip[base + d0] = a0;          snip[base + d1] = a1;
    snip_bf[base + d0] = f2bf(a0); snip_bf[base + d1] = f2bf(a1);
}

// ---------------------------------------------------------------------------
// out[b,q] = exp(-|| normalize(audio*g_s) - normalize(snippet*g_u) ||)
// ---------------------------------------------------------------------------
__global__ __launch_bounds__(256) void final_kernel(
    const float* __restrict__ audio, const float* __restrict__ snip,
    const float* __restrict__ g_u,   const float* __restrict__ g_s,
    float* __restrict__ out)
{
    __shared__ float red[3][8];
    const int q = blockIdx.x, b = blockIdx.y, tid = threadIdx.x;
    const int lane = tid & 31, wid = tid >> 5;
    const unsigned long long base = (unsigned long long)(b * LQ_ + q) * D_;

    float uu = 0.f, ss = 0.f, us = 0.f;
    #pragma unroll
    for (int p = 0; p < 2; ++p) {
        const int d = tid + p * 256;
        const float u = audio[base + d] * g_s[base + d];
        const float s = snip[base + d] * g_u[base + d];
        uu = fmaf(u, u, uu);
        ss = fmaf(s, s, ss);
        us = fmaf(u, s, us);
    }
    #pragma unroll
    for (int off = 16; off > 0; off >>= 1) {
        uu += __shfl_xor(uu, off);
        ss += __shfl_xor(ss, off);
        us += __shfl_xor(us, off);
    }
    if (lane == 0) { red[0][wid] = uu; red[1][wid] = ss; red[2][wid] = us; }
    __syncthreads();
    if (tid == 0) {
        float tuu = 0.f, tss = 0.f, tus = 0.f;
        #pragma unroll
        for (int i = 0; i < 8; ++i) { tuu += red[0][i]; tss += red[1][i]; tus += red[2][i]; }
        const float nu = fmaxf(sqrtf(tuu), 1e-12f);
        const float ns = fmaxf(sqrtf(tss), 1e-12f);
        float t = tuu / (nu * nu) + tss / (ns * ns) - 2.0f * tus / (nu * ns);
        out[b * LQ_ + q] = __expf(-sqrtf(fmaxf(t, 0.0f)));
    }
}

// ---------------------------------------------------------------------------
extern "C" void kernel_launch(void* const* d_in, const int* in_sizes, int n_in,
                              void* d_out, int out_size, void* d_ws, size_t ws_size,
                              hipStream_t stream) {
    (void)in_sizes; (void)n_in; (void)out_size; (void)ws_size;

    const float* audio     = (const float*)d_in[0];
    const float* text      = (const float*)d_in[1];
    const int*   audio_len = (const int*)  d_in[2];
    const int*   text_len  = (const int*)  d_in[3];
    const float* h2attn_w  = (const float*)d_in[4];
    const float* h2attn_b  = (const float*)d_in[5];
    const float* vvec      = (const float*)d_in[6];
    const float* fc_u_w    = (const float*)d_in[7];
    const float* fc_u_b    = (const float*)d_in[8];
    const float* fc_s_w    = (const float*)d_in[9];
    const float* fc_s_b    = (const float*)d_in[10];
    float* out = (float*)d_out;

    const int nA = B_ * LQ_ * D_;    // 4,194,304
    const int nT = B_ * LKV_ * D_;   //   262,144
    const int nW = D_ * D_;          //   262,144

    char* w = (char*)d_ws;
    size_t off = 0;
    auto alloc = [&](size_t bytes) -> void* {
        void* p = w + off;
        off += (bytes + 255) & ~(size_t)255;
        return p;
    };
    unsigned short* audio_bf = (unsigned short*)alloc((size_t)nA * 2);
    unsigned short* text_bf  = (unsigned short*)alloc((size_t)nT * 2);
    unsigned short* wq_bf    = (unsigned short*)alloc((size_t)nW * 2);
    unsigned short* wkv_bf   = (unsigned short*)alloc((size_t)nW * 2);
    unsigned short* fcu_bf   = (unsigned short*)alloc((size_t)nW * 2);
    unsigned short* fcs_bf   = (unsigned short*)alloc((size_t)nW * 2);
    float* qa   = (float*)alloc((size_t)nA * 4);   // reused as g_u later
    float* ka   = (float*)alloc((size_t)nT * 4);
    float* attn = (float*)alloc((size_t)B_ * LQ_ * LKV_ * 4);
    float* snip = (float*)alloc((size_t)nA * 4);
    unsigned short* snip_bf = (unsigned short*)alloc((size_t)nA * 2);
    float* gs   = (float*)alloc((size_t)nA * 4);

    // 1) precision conversion (one pass; whole problem is L2-resident after)
    cvt_bf16<<<(nA + 255) / 256, 256, 0, stream>>>(audio, audio_bf, nA);
    cvt_bf16<<<(nT + 255) / 256, 256, 0, stream>>>(text, text_bf, nT);
    cvt_bf16<<<(nW + 255) / 256, 256, 0, stream>>>(fc_u_w, fcu_bf, nW);
    cvt_bf16<<<(nW + 255) / 256, 256, 0, stream>>>(fc_s_w, fcs_bf, nW);
    split_h2attn<<<(nW + 255) / 256, 256, 0, stream>>>(h2attn_w, wq_bf, wkv_bf);

    const unsigned long long sA = (unsigned long long)LQ_ * D_;
    const unsigned long long sT = (unsigned long long)LKV_ * D_;
    dim3 gBig(LQ_ / 128, D_ / 128, B_);   // (4, 4, 16)
    dim3 gKa(1, D_ / 64, B_);             // (1, 8, 16)

    // 2) qa = audio @ Wq^T ; ka = text @ Wkv^T + h2attn_b
    gemm_bf16_wmma_async<0><<<gBig, 256, 0, stream>>>(audio_bf, wq_bf, nullptr, qa,
                                                      LQ_, D_, D_, sA, sA);
    gemm_bf16_wmma_m32<<<gKa, 256, 0, stream>>>(text_bf, wkv_bf, h2attn_b, ka,
                                                D_, D_, sT, sT);
    // 3) additive-attention scores + masked softmax
    score_softmax<<<dim3(LQ_, B_), 256, 0, stream>>>(qa, ka, vvec,
                                                     audio_len, text_len, attn);
    // 4) snippet = attn @ text
    snippet_kernel<<<dim3(LQ_, B_), 256, 0, stream>>>(attn, text, snip, snip_bf);
    // 5) gates: g_u = sigmoid(audio @ fc_u^T), g_s = sigmoid(snippet @ fc_s^T)
    gemm_bf16_wmma_async<2><<<gBig, 256, 0, stream>>>(audio_bf, fcu_bf, fc_u_b, qa,
                                                      LQ_, D_, D_, sA, sA);
    gemm_bf16_wmma_async<2><<<gBig, 256, 0, stream>>>(snip_bf, fcs_bf, fc_s_b, gs,
                                                      LQ_, D_, D_, sA, sA);
    // 6) fused normalize + distance + exp
    final_kernel<<<dim3(LQ_, B_), 256, 0, stream>>>(audio, snip, qa, gs, out);
}
